// EntityExtractor_55473797595940
// MI455X (gfx1250) — compile-verified, matrix operand order checked by reading, not run on hardware
//
#include <hip/hip_runtime.h>
#include <hip/hip_bf16.h>
#include <math.h>

typedef __bf16 bf16_t;
typedef __attribute__((ext_vector_type(16))) __bf16 v16bf;
typedef __attribute__((ext_vector_type(8)))  float  v8f;

#define NB 8
#define NS 4096
#define ND 1024
#define NK 20
#define NT 10
#define MINSC 0.4f
#define LNEPS 1e-5f
#define MAXRUNS (NS / 2)

// ---------- helpers ----------

__device__ __forceinline__ float wave_sum32(float v) {
#pragma unroll
  for (int off = 16; off > 0; off >>= 1) v += __shfl_xor(v, off, 32);
  return v;
}

__device__ __forceinline__ float gelu_exact(float x) {
  return 0.5f * x * (1.0f + erff(x * 0.70710678118654752440f));
}

// Fragment loader for 16x16x32 bf16 WMMA (ISA 7.12.2 layout).
// rowp points at the start of this lane's row (A: m = lane&15, B: n = lane&15).
__device__ __forceinline__ v16bf load_frag_row(const bf16_t* __restrict__ rowp,
                                               int k0, int lane) {
  const bf16_t* p = rowp + k0 + ((lane & 16) ? 8 : 0);
  v16bf v;
#pragma unroll
  for (int i = 0; i < 8; ++i) v[i] = p[i];
#pragma unroll
  for (int i = 0; i < 8; ++i) v[8 + i] = p[16 + i];
  return v;
}

// ---------- kernel 0: transpose f32 [ND,ND] -> bf16 [ND,ND] (out[j][i]=in[i][j]) ----------

__global__ __launch_bounds__(256) void k_transpose_bf16(const float* __restrict__ in,
                                                        bf16_t* __restrict__ out) {
  int j = blockIdx.x * 16 + (threadIdx.x & 15);
  int i = blockIdx.y * 16 + (threadIdx.x >> 4);
  out[(size_t)j * ND + i] = (bf16_t)in[(size_t)i * ND + j];
}

// ---------- kernel 1: fused score head, M-tile = 32 ----------
// Each B fragment feeds two WMMAs (two 16-row sub-tiles) -> half the L2 weight
// traffic and two independent accumulator chains.

__global__ __launch_bounds__(256) void k_score_head(
    const float* __restrict__ hs, const float* __restrict__ mask,
    const bf16_t* __restrict__ w1t, const float* __restrict__ b1,
    const float* __restrict__ g, const float* __restrict__ be,
    const float* __restrict__ w2, const float* __restrict__ b2,
    float* __restrict__ scores_out) {
  __shared__ bf16_t As[32 * ND];   // 64 KB
  __shared__ float  Hs[32 * ND];   // 128 KB
  const int tid = threadIdx.x;
  const int lane = tid & 31, wave = tid >> 5;
  const long long row0 = (long long)blockIdx.x * 32;

  // stage A tile (32 tokens x ND), convert f32 -> bf16
  for (int idx = tid; idx < 32 * ND; idx += 256) {
    int m = idx >> 10, d = idx & (ND - 1);
    As[idx] = (bf16_t)hs[(row0 + m) * ND + d];
  }
  __syncthreads();

  const bf16_t* arow0 = &As[(size_t)(lane & 15) * ND];
  const bf16_t* arow1 = &As[(size_t)((lane & 15) + 16) * ND];
#pragma unroll 1
  for (int j = 0; j < 8; ++j) {
    const int n0 = (wave * 8 + j) * 16;
    const bf16_t* brow = &w1t[(size_t)(n0 + (lane & 15)) * ND];
    if (j < 7) __builtin_prefetch(brow + 16 * ND, 0, 1);  // next n-tile row
    v8f acc0 = {}, acc1 = {};
#pragma unroll 4
    for (int k0 = 0; k0 < ND; k0 += 32) {
      v16bf b  = load_frag_row(brow, k0, lane);
      v16bf a0 = load_frag_row(arow0, k0, lane);
      v16bf a1 = load_frag_row(arow1, k0, lane);
      acc0 = __builtin_amdgcn_wmma_f32_16x16x32_bf16(false, a0, false, b,
                                                     (short)0, acc0, false, false);
      acc1 = __builtin_amdgcn_wmma_f32_16x16x32_bf16(false, a1, false, b,
                                                     (short)0, acc1, false, false);
    }
    const int n = n0 + (lane & 15);
    const int mo = (lane & 16) ? 8 : 0;
    const float bias = b1[n];
#pragma unroll
    for (int r = 0; r < 8; ++r) {
      Hs[(r + mo) * ND + n]      = acc0[r] + bias;
      Hs[(16 + r + mo) * ND + n] = acc1[r] + bias;
    }
  }
  __syncthreads();

  // per-row LayerNorm + GELU + dot(w2), 4 rows per wave
  for (int m = wave * 4; m < wave * 4 + 4; ++m) {
    float s = 0.f, s2 = 0.f;
    for (int c = lane; c < ND; c += 32) {
      float v = Hs[m * ND + c];
      s += v; s2 += v * v;
    }
    s = wave_sum32(s); s2 = wave_sum32(s2);
    const float mu = s * (1.0f / ND);
    const float var = s2 * (1.0f / ND) - mu * mu;
    const float inv = rsqrtf(var + LNEPS);
    float accd = 0.f;
    for (int c = lane; c < ND; c += 32) {
      float x = (Hs[m * ND + c] - mu) * inv * g[c] + be[c];
      accd += gelu_exact(x) * w2[c];
    }
    accd = wave_sum32(accd);
    if (lane == 0) {
      long long tok = row0 + m;
      float sc = 1.0f / (1.0f + expf(-(accd + b2[0])));
      scores_out[tok] = sc * mask[tok];
    }
  }
}

// ---------- kernel 2: wave-parallel run-length span detection + stable top-K ----------
// One wave per batch row. Per 32-token chunk: ballot -> active mask, shuffle
// prefix-sum -> run score sums, uniform bit-scan -> runs (with cross-chunk carry).

__global__ __launch_bounds__(32) void k_spans(
    const float* __restrict__ scores, const float* __restrict__ mask,
    int* __restrict__ sel_st, int* __restrict__ sel_en,
    float* __restrict__ sel_cn, int* __restrict__ sel_valid,
    float* __restrict__ o_topv, float* __restrict__ o_st,
    float* __restrict__ o_en, float* __restrict__ o_valid) {
  __shared__ float Lv[MAXRUNS];   // run average score
  __shared__ int   Lse[MAXRUNS];  // packed start | (end << 16)
  const int b = blockIdx.x;
  const int lane = threadIdx.x;
  const float* sc = scores + (size_t)b * NS;
  const float* mk = mask + (size_t)b * NS;

  int nc = 0;                 // uniform across lanes
  int runStart = -1, runCnt = 0;
  float runSum = 0.f;

  auto commit = [&](int cs_, int ce_, float sum_, int cnt_) {
    float avg = sum_ / (float)cnt_;
    if (avg >= MINSC && nc < MAXRUNS) {
      if (lane == 0) { Lv[nc] = avg; Lse[nc] = cs_ | (ce_ << 16); }
      ++nc;
    }
  };

  for (int c = 0; c < NS; c += 32) {
    const int s = c + lane;
    const float scv = sc[s];
    const bool act = (mk[s] > 0.f) && (scv >= MINSC);
    const unsigned m = (unsigned)(__ballot(act) & 0xffffffffull);
    // inclusive prefix sum of active scores
    float ps = act ? scv : 0.f;
#pragma unroll
    for (int off = 1; off < 32; off <<= 1) {
      float t = __shfl_up(ps, off, 32);
      if (lane >= off) ps += t;
    }
    auto S = [&](int i) -> float { return __shfl(ps, i, 32); };

    int pos = 0;
    if (runStart >= 0) {
      if (m & 1u) {
        const int len0 = (m == 0xffffffffu) ? 32 : __builtin_ctz(~m);
        runSum += S(len0 - 1);
        runCnt += len0;
        pos = len0;
        if (len0 < 32) { commit(runStart, c + len0 - 1, runSum, runCnt); runStart = -1; }
      } else {
        commit(runStart, c - 1, runSum, runCnt);
        runStart = -1;
      }
    }
    while (pos < 32) {
      const unsigned rem = m >> pos;
      if (!rem) break;
      const int st = pos + __builtin_ctz(rem);
      const unsigned sh = m >> st;
      const int len = (sh == 0xffffffffu) ? 32 : __builtin_ctz(~sh);
      const float sum = S(st + len - 1) - ((st > 0) ? S(st - 1) : 0.f);
      if (st + len == 32) {  // run reaches chunk end -> carry
        runStart = c + st; runSum = sum; runCnt = len;
        pos = 32;
      } else {
        commit(c + st, c + st + len - 1, sum, len);
        pos = st + len;
      }
    }
  }
  if (runStart >= 0) commit(runStart, NS - 1, runSum, runCnt);
  __syncthreads();

  // stable top-K: wave-parallel argmax, ties -> lowest index (matches lax.top_k)
  unsigned long long chosen = 0;  // per-lane ownership: candidate i -> lane i&31, bit i>>5
  for (int k = 0; k < NK; ++k) {
    float bv = -1.f;
    int bi = 0x7fffffff;
    for (int t = 0; t < MAXRUNS / 32; ++t) {
      const int i = (t << 5) + lane;
      if (i < nc && !((chosen >> t) & 1ull)) {
        float v = Lv[i];
        if (v > bv) { bv = v; bi = i; }  // ascending i => strict > keeps lowest
      }
    }
#pragma unroll
    for (int off = 16; off > 0; off >>= 1) {
      float ov = __shfl_xor(bv, off, 32);
      int oi = __shfl_xor(bi, off, 32);
      if (ov > bv || (ov == bv && oi < bi)) { bv = ov; bi = oi; }
    }
    const int o = b * NK + k;
    if (bv >= MINSC) {
      if ((bi & 31) == lane) chosen |= 1ull << (bi >> 5);
      const int se = Lse[bi];
      const int cs_ = se & 0xffff, ce_ = (se >> 16) & 0xffff;
      if (lane == 0) {
        const int cnt = ce_ - cs_ + 1;
        sel_st[o] = cs_; sel_en[o] = ce_; sel_cn[o] = (float)cnt; sel_valid[o] = 1;
        o_topv[o] = bv; o_st[o] = (float)cs_; o_en[o] = (float)ce_; o_valid[o] = 1.0f;
      }
    } else if (lane == 0) {
      sel_st[o] = 0; sel_en[o] = 0; sel_cn[o] = 1.0f; sel_valid[o] = 0;
      o_topv[o] = -1.0f; o_st[o] = 0.f; o_en[o] = 0.f; o_valid[o] = 0.f;
    }
  }
}

// ---------- kernel 3: span means (direct row sums), emit bf16 ----------

__global__ __launch_bounds__(256) void k_span_mean(
    const float* __restrict__ hs, const int* __restrict__ sel_st,
    const int* __restrict__ sel_en, const float* __restrict__ sel_cn,
    const int* __restrict__ sel_valid, bf16_t* __restrict__ sm) {
  const int span = blockIdx.x;  // b*NK + k
  const int b = span / NK;
  const int st = sel_st[span], en = sel_en[span];
  const float cn = sel_cn[span];
  const int valid = sel_valid[span];
  for (int c = threadIdx.x; c < ND; c += 256) {
    float s = 0.f;
    if (valid)
      for (int r = st; r <= en; ++r) s += hs[((size_t)b * NS + r) * ND + c];
    sm[(size_t)span * ND + c] = (bf16_t)(valid ? s / cn : 0.0f);
  }
}

// ---------- kernel 4: fused enhanced head (GEMM->LN->GELU->GEMM, masked) ----------

__global__ __launch_bounds__(256) void k_enhance(
    const bf16_t* __restrict__ sm, const bf16_t* __restrict__ w1t,
    const float* __restrict__ b1, const float* __restrict__ g,
    const float* __restrict__ be, const bf16_t* __restrict__ w2t,
    const float* __restrict__ b2, const int* __restrict__ sel_valid,
    float* __restrict__ out_enh, bf16_t* __restrict__ enh_bf) {
  __shared__ bf16_t As[16 * ND];  // input tile, later reused for GELU tile
  __shared__ float  Hs[16 * ND];
  const int tid = threadIdx.x;
  const int lane = tid & 31, wave = tid >> 5;
  const int row0 = blockIdx.x * 16;

  for (int idx = tid; idx < 16 * ND; idx += 256)
    As[idx] = sm[(size_t)row0 * ND + idx];
  __syncthreads();

  const bf16_t* arow = &As[(size_t)(lane & 15) * ND];
#pragma unroll 1
  for (int j = 0; j < 8; ++j) {
    const int n0 = (wave * 8 + j) * 16;
    const bf16_t* brow = &w1t[(size_t)(n0 + (lane & 15)) * ND];
    v8f acc = {};
#pragma unroll 4
    for (int k0 = 0; k0 < ND; k0 += 32) {
      v16bf a = load_frag_row(arow, k0, lane);
      v16bf b = load_frag_row(brow, k0, lane);
      acc = __builtin_amdgcn_wmma_f32_16x16x32_bf16(false, a, false, b,
                                                    (short)0, acc, false, false);
    }
    const int n = n0 + (lane & 15);
    const int mo = (lane & 16) ? 8 : 0;
    const float bias = b1[n];
#pragma unroll
    for (int r = 0; r < 8; ++r) Hs[(r + mo) * ND + n] = acc[r] + bias;
  }
  __syncthreads();

  // LN + GELU, repack bf16 into As (input tile is dead)
  for (int m = wave * 2; m < wave * 2 + 2; ++m) {
    float s = 0.f, s2 = 0.f;
    for (int c = lane; c < ND; c += 32) {
      float v = Hs[m * ND + c];
      s += v; s2 += v * v;
    }
    s = wave_sum32(s); s2 = wave_sum32(s2);
    const float mu = s * (1.0f / ND);
    const float inv = rsqrtf(s2 * (1.0f / ND) - mu * mu + LNEPS);
    for (int c = lane; c < ND; c += 32) {
      float x = (Hs[m * ND + c] - mu) * inv * g[c] + be[c];
      As[m * ND + c] = (bf16_t)gelu_exact(x);
    }
  }
  __syncthreads();

  // second GEMM: gelu(LN) @ w2  (+ bias, masked by validity)
#pragma unroll 1
  for (int j = 0; j < 8; ++j) {
    const int n0 = (wave * 8 + j) * 16;
    const bf16_t* brow = &w2t[(size_t)(n0 + (lane & 15)) * ND];
    v8f acc = {};
#pragma unroll 4
    for (int k0 = 0; k0 < ND; k0 += 32) {
      v16bf a = load_frag_row(arow, k0, lane);
      v16bf b = load_frag_row(brow, k0, lane);
      acc = __builtin_amdgcn_wmma_f32_16x16x32_bf16(false, a, false, b,
                                                    (short)0, acc, false, false);
    }
    const int n = n0 + (lane & 15);
    const int mo = (lane & 16) ? 8 : 0;
    const float bias = b2[n];
#pragma unroll
    for (int r = 0; r < 8; ++r) {
      const int row = row0 + r + mo;
      const float vm = sel_valid[row] ? 1.0f : 0.0f;
      const float v = (acc[r] + bias) * vm;
      out_enh[(size_t)row * ND + n] = v;
      enh_bf[(size_t)row * ND + n] = (bf16_t)v;
    }
  }
}

// ---------- kernel 5: type head (GEMM->LN->GELU->small linear->softmax/argmax) ----------

__global__ __launch_bounds__(256) void k_type_head(
    const bf16_t* __restrict__ enh_bf, const bf16_t* __restrict__ w1t,
    const float* __restrict__ b1, const float* __restrict__ g,
    const float* __restrict__ be, const float* __restrict__ w2,
    const float* __restrict__ b2, float* __restrict__ out_logits,
    float* __restrict__ out_probs, float* __restrict__ out_types) {
  __shared__ bf16_t As[16 * ND];
  __shared__ float  Hs[16 * ND];
  __shared__ float  Ls[16 * NT];
  const int tid = threadIdx.x;
  const int lane = tid & 31, wave = tid >> 5;
  const int row0 = blockIdx.x * 16;

  for (int idx = tid; idx < 16 * ND; idx += 256)
    As[idx] = enh_bf[(size_t)row0 * ND + idx];
  __syncthreads();

  const bf16_t* arow = &As[(size_t)(lane & 15) * ND];
#pragma unroll 1
  for (int j = 0; j < 8; ++j) {
    const int n0 = (wave * 8 + j) * 16;
    const bf16_t* brow = &w1t[(size_t)(n0 + (lane & 15)) * ND];
    v8f acc = {};
#pragma unroll 4
    for (int k0 = 0; k0 < ND; k0 += 32) {
      v16bf a = load_frag_row(arow, k0, lane);
      v16bf b = load_frag_row(brow, k0, lane);
      acc = __builtin_amdgcn_wmma_f32_16x16x32_bf16(false, a, false, b,
                                                    (short)0, acc, false, false);
    }
    const int n = n0 + (lane & 15);
    const int mo = (lane & 16) ? 8 : 0;
    const float bias = b1[n];
#pragma unroll
    for (int r = 0; r < 8; ++r) Hs[(r + mo) * ND + n] = acc[r] + bias;
  }
  __syncthreads();

  // LN + GELU in place (f32)
  for (int m = wave * 2; m < wave * 2 + 2; ++m) {
    float s = 0.f, s2 = 0.f;
    for (int c = lane; c < ND; c += 32) {
      float v = Hs[m * ND + c];
      s += v; s2 += v * v;
    }
    s = wave_sum32(s); s2 = wave_sum32(s2);
    const float mu = s * (1.0f / ND);
    const float inv = rsqrtf(s2 * (1.0f / ND) - mu * mu + LNEPS);
    for (int c = lane; c < ND; c += 32) {
      float x = (Hs[m * ND + c] - mu) * inv * g[c] + be[c];
      Hs[m * ND + c] = gelu_exact(x);
    }
  }
  __syncthreads();

  // logits: 160 (row, type) pairs
  for (int p = tid; p < 16 * NT; p += 256) {
    const int m = p / NT, t = p % NT;
    float s = 0.f;
    for (int h = 0; h < ND; ++h) s += Hs[m * ND + h] * w2[h * NT + t];
    s += b2[t];
    Ls[p] = s;
    out_logits[(size_t)(row0 + m) * NT + t] = s;
  }
  __syncthreads();

  if (tid < 16) {
    const int m = tid;
    float mx = Ls[m * NT]; int ai = 0;
    for (int t = 1; t < NT; ++t)
      if (Ls[m * NT + t] > mx) { mx = Ls[m * NT + t]; ai = t; }
    float ex[NT], se = 0.f;
    for (int t = 0; t < NT; ++t) { ex[t] = expf(Ls[m * NT + t] - mx); se += ex[t]; }
    const float rse = 1.0f / se;
    for (int t = 0; t < NT; ++t)
      out_probs[(size_t)(row0 + m) * NT + t] = ex[t] * rse;
    out_types[row0 + m] = (float)ai;
  }
}

// ---------- launch ----------

static inline size_t alignup256(size_t x) { return (x + 255) & ~(size_t)255; }

extern "C" void kernel_launch(void* const* d_in, const int* in_sizes, int n_in,
                              void* d_out, int out_size, void* d_ws, size_t ws_size,
                              hipStream_t stream) {
  (void)in_sizes; (void)n_in; (void)out_size; (void)ws_size;
  const float* hs    = (const float*)d_in[0];
  const float* mask  = (const float*)d_in[1];
  const float* sc_w1 = (const float*)d_in[2];
  const float* sc_b1 = (const float*)d_in[3];
  const float* sc_g  = (const float*)d_in[4];
  const float* sc_be = (const float*)d_in[5];
  const float* sc_w2 = (const float*)d_in[6];
  const float* sc_b2 = (const float*)d_in[7];
  const float* en_w1 = (const float*)d_in[8];
  const float* en_b1 = (const float*)d_in[9];
  const float* en_g  = (const float*)d_in[10];
  const float* en_be = (const float*)d_in[11];
  const float* en_w2 = (const float*)d_in[12];
  const float* en_b2 = (const float*)d_in[13];
  const float* ty_w1 = (const float*)d_in[14];
  const float* ty_b1 = (const float*)d_in[15];
  const float* ty_g  = (const float*)d_in[16];
  const float* ty_be = (const float*)d_in[17];
  const float* ty_w2 = (const float*)d_in[18];
  const float* ty_b2 = (const float*)d_in[19];

  // workspace layout
  char* w = (char*)d_ws;
  size_t o = 0;
  bf16_t* scw1t = (bf16_t*)(w + o); o += alignup256((size_t)ND * ND * 2);
  bf16_t* enw1t = (bf16_t*)(w + o); o += alignup256((size_t)ND * ND * 2);
  bf16_t* tyw1t = (bf16_t*)(w + o); o += alignup256((size_t)ND * ND * 2);
  bf16_t* enw2t = (bf16_t*)(w + o); o += alignup256((size_t)ND * ND * 2);
  int*   sel_st    = (int*)  (w + o); o += alignup256((size_t)NB * NK * 4);
  int*   sel_en    = (int*)  (w + o); o += alignup256((size_t)NB * NK * 4);
  float* sel_cn    = (float*)(w + o); o += alignup256((size_t)NB * NK * 4);
  int*   sel_valid = (int*)  (w + o); o += alignup256((size_t)NB * NK * 4);
  bf16_t* sm_bf  = (bf16_t*)(w + o); o += alignup256((size_t)NB * NK * ND * 2);
  bf16_t* enh_bf = (bf16_t*)(w + o); o += alignup256((size_t)NB * NK * ND * 2);

  // output layout (reference tuple order, flattened)
  float* out = (float*)d_out;
  float* o_enh    = out;                         // [8,20,1024]
  float* o_logits = o_enh + NB * NK * ND;        // [8,20,10]
  float* o_probs  = o_logits + NB * NK * NT;     // [8,20,10]
  float* o_types  = o_probs + NB * NK * NT;      // [8,20]
  float* o_topv   = o_types + NB * NK;           // [8,20]
  float* o_st     = o_topv + NB * NK;            // [8,20]
  float* o_en     = o_st + NB * NK;              // [8,20]
  float* o_valid  = o_en + NB * NK;              // [8,20]
  float* o_scores = o_valid + NB * NK;           // [8,4096]

  dim3 tgrid(ND / 16, ND / 16);
  k_transpose_bf16<<<tgrid, 256, 0, stream>>>(sc_w1, scw1t);
  k_transpose_bf16<<<tgrid, 256, 0, stream>>>(en_w1, enw1t);
  k_transpose_bf16<<<tgrid, 256, 0, stream>>>(ty_w1, tyw1t);
  k_transpose_bf16<<<tgrid, 256, 0, stream>>>(en_w2, enw2t);

  k_score_head<<<(NB * NS) / 32, 256, 0, stream>>>(
      hs, mask, scw1t, sc_b1, sc_g, sc_be, sc_w2, sc_b2, o_scores);

  k_spans<<<NB, 32, 0, stream>>>(o_scores, mask,
                                 sel_st, sel_en, sel_cn, sel_valid,
                                 o_topv, o_st, o_en, o_valid);

  k_span_mean<<<NB * NK, 256, 0, stream>>>(hs, sel_st, sel_en, sel_cn, sel_valid, sm_bf);

  k_enhance<<<(NB * NK) / 16, 256, 0, stream>>>(
      sm_bf, enw1t, en_b1, en_g, en_be, enw2t, en_b2, sel_valid, o_enh, enh_bf);

  k_type_head<<<(NB * NK) / 16, 256, 0, stream>>>(
      enh_bf, tyw1t, ty_b1, ty_g, ty_be, ty_w2, ty_b2, o_logits, o_probs, o_types);
}
